// ImprovedPlasticNet_2336462209822
// MI455X (gfx1250) — matrix operand order repaired and checked by reading, not compiled, not run on hardware
//
#include <hip/hip_runtime.h>

// ---------------------------------------------------------------------------
// MI455X (gfx1250, wave32) bf16-WMMA GEMM chain for ImprovedPlasticNet.
//   h0 = relu(x @ in_w^T + in_b)
//   5x: h = relu(h @ (W*adj)); credit += h_prev^T @ h / 1024
//   out = h @ out_w^T + out_b
// Inner loop: v_wmma_f32_16x16x32_bf16, LDS ping-pong, 1 barrier / K-step.
// Contiguous operands staged via GLOBAL_LOAD_ASYNC_TO_LDS_B128 (ASYNCcnt);
// scatter operands use a register-prefetch pipeline.
// ---------------------------------------------------------------------------

typedef __bf16 bf16;
typedef __attribute__((ext_vector_type(8)))  bf16  v8bf;
typedef __attribute__((ext_vector_type(16))) bf16  v16bf;
typedef __attribute__((ext_vector_type(8)))  float v8f;
typedef int v4i_vs __attribute__((__vector_size__(16)));   // matches builtin param

#define AS1 __attribute__((address_space(1)))
#define AS3 __attribute__((address_space(3)))
#define GPTR(p) ((AS1 v4i_vs*)(p))
#define LPTR(p) ((AS3 v4i_vs*)(p))

#if defined(__has_builtin)
#  if __has_builtin(__builtin_amdgcn_global_load_async_to_lds_b128) && \
      __has_builtin(__builtin_amdgcn_s_wait_asynccnt)
#    define HAVE_ASYNC_LDS 1
#  endif
#endif
#ifndef HAVE_ASYNC_LDS
#  define HAVE_ASYNC_LDS 0
#endif

__device__ __forceinline__ bf16 to_bf16(float f) {
  union { float f; unsigned u; } x; x.f = f;
  unsigned r = (x.u + 0x7FFFu + ((x.u >> 16) & 1u)) >> 16;   // RNE
  union { unsigned short s; bf16 b; } y; y.s = (unsigned short)r;
  return y.b;
}

__global__ void cvt_f32_bf16(const float* __restrict__ s, bf16* __restrict__ d, int n) {
  int i = blockIdx.x * blockDim.x + threadIdx.x;
  if (i < n) d[i] = to_bf16(s[i]);
}

__global__ void mask_mul_bf16(const float* __restrict__ w, const float* __restrict__ m,
                              bf16* __restrict__ d, int n) {
  int i = blockIdx.x * blockDim.x + threadIdx.x;
  if (i < n) d[i] = to_bf16(w[i] * m[i]);
}

__global__ void zero_f32(float* __restrict__ p, int n) {
  int i = blockIdx.x * blockDim.x + threadIdx.x;
  if (i < n) p[i] = 0.f;
}

// ---------------------------------------------------------------------------
// Tiled GEMM: D = A_eff(MxK) * B_eff(KxN). Macro tile 64(M) x 128(N), BK=32.
// 256 threads = 8 wave32s arranged 2(M) x 4(N); each wave owns a 32x32 tile.
// TA: A_eff[m][k] = TA ? A[k*lda+m] : A[m*lda+k]
// TB: B_eff[k][n] = TB ? B[n*ldb+k] : B[k*ldb+n]
// OUTM: 0 = store f32 (+bias), 1 = store bf16 (+bias), 2 = outF += alpha*D
// ---------------------------------------------------------------------------
#define ASTR 48   // LDS row stride in bf16 (96B: keeps 16B/32B frags aligned)

template <int TA, int TB, int RELU, int OUTM>
__global__ __launch_bounds__(256) void gemm_bf16_wmma(
    const bf16* __restrict__ A, int lda,
    const bf16* __restrict__ B, int ldb,
    const float* __restrict__ bias,
    float* __restrict__ outF,
    bf16* __restrict__ outH,
    float alpha, int N, int K)
{
  __shared__ __align__(16) bf16 As[2][64  * ASTR];
  __shared__ __align__(16) bf16 Bs[2][128 * ASTR];   // stored as Bs[n][k]

  constexpr bool ASY_A = HAVE_ASYNC_LDS && (TA == 0);
  constexpr bool ASY_B = HAVE_ASYNC_LDS && (TB == 1);

  const int t    = threadIdx.x;
  const int lane = t & 31;
  const int wave = t >> 5;
  const int wm   = (wave & 1) * 32;     // wave M offset in macro tile
  const int wn   = (wave >> 1) * 32;    // wave N offset
  const int fr   = lane & 15;
  const int hq   = lane >> 4;           // half-wave index
  const int m0   = blockIdx.y * 64;
  const int n0   = blockIdx.x * 128;

  v8f acc00 = {}, acc01 = {}, acc10 = {}, acc11 = {};
  v8bf pa, pb0, pb1;                    // register prefetch buffers (scatter paths)

  // ---- async global->LDS staging for contiguous operands (ASYNCcnt path)
  auto issue_async = [&](int kt, int bufi) {
#if HAVE_ASYNC_LDS
    if (ASY_A) {
      int m = t >> 2, k8 = (t & 3) << 3;
      __builtin_amdgcn_global_load_async_to_lds_b128(
          GPTR(A + (size_t)(m0 + m) * lda + (kt + k8)),
          LPTR(&As[bufi][m * ASTR + k8]), 0, 0);
    }
    if (ASY_B) {
      int n = t >> 2, k8 = (t & 3) << 3;
      __builtin_amdgcn_global_load_async_to_lds_b128(
          GPTR(B + (size_t)(n0 + n) * ldb + (kt + k8)),
          LPTR(&Bs[bufi][n * ASTR + k8]), 0, 0);
      __builtin_amdgcn_global_load_async_to_lds_b128(
          GPTR(B + (size_t)(n0 + n + 64) * ldb + (kt + k8)),
          LPTR(&Bs[bufi][(n + 64) * ASTR + k8]), 0, 0);
    }
#endif
  };
  auto wait_async = [&]() {
#if HAVE_ASYNC_LDS
    if (ASY_A || ASY_B) __builtin_amdgcn_s_wait_asynccnt(0);
#endif
  };

  // ---- register fetch for operands not handled by the async engine
  auto fetch = [&](int kt) {
    if (TA == 0) {
      if (!ASY_A) {
        int m = t >> 2, k8 = (t & 3) << 3;
        pa = *(const v8bf*)(A + (size_t)(m0 + m) * lda + (kt + k8));
      }
    } else {
      int k = t >> 3, m8 = (t & 7) << 3;
      pa = *(const v8bf*)(A + (size_t)(kt + k) * lda + (m0 + m8));
    }
    if (TB == 1) {
      if (!ASY_B) {
        int n = t >> 2, k8 = (t & 3) << 3;
        pb0 = *(const v8bf*)(B + (size_t)(n0 + n)      * ldb + (kt + k8));
        pb1 = *(const v8bf*)(B + (size_t)(n0 + n + 64) * ldb + (kt + k8));
      }
    } else {
      int k = t >> 3, n8 = (t & 7) << 3;
      pb0 = *(const v8bf*)(B + (size_t)(kt + k) * ldb + (n0 + n8));
      pb1 = *(const v8bf*)(B + (size_t)(kt + k) * ldb + (n0 + n8 + 64));
    }
  };

  // ---- warm L2 two K-tiles ahead (global_prefetch_b8)
  auto prefetch = [&](int kt) {
    if (TA == 0) {
      int m = t >> 2, k8 = (t & 3) << 3;
      __builtin_prefetch(A + (size_t)(m0 + m) * lda + (kt + k8), 0, 1);
    } else {
      int k = t >> 3, m8 = (t & 7) << 3;
      __builtin_prefetch(A + (size_t)(kt + k) * lda + (m0 + m8), 0, 1);
    }
    if (TB == 1) {
      int n = t >> 2, k8 = (t & 3) << 3;
      __builtin_prefetch(B + (size_t)(n0 + n) * ldb + (kt + k8), 0, 1);
    } else {
      int k = t >> 3, n8 = (t & 7) << 3;
      __builtin_prefetch(B + (size_t)(kt + k) * ldb + (n0 + n8), 0, 1);
    }
  };

  // ---- stage register-path operands into LDS buffer `bufi`
  auto stage = [&](int bufi) {
    if (TA == 0) {
      if (!ASY_A) {
        int m = t >> 2, k8 = (t & 3) << 3;
        *(v8bf*)(&As[bufi][m * ASTR + k8]) = pa;
      }
    } else {
      int k = t >> 3, m8 = (t & 7) << 3;
      #pragma unroll
      for (int i = 0; i < 8; ++i) As[bufi][(m8 + i) * ASTR + k] = pa[i];
    }
    if (TB == 1) {
      if (!ASY_B) {
        int n = t >> 2, k8 = (t & 3) << 3;
        *(v8bf*)(&Bs[bufi][n * ASTR + k8])        = pb0;
        *(v8bf*)(&Bs[bufi][(n + 64) * ASTR + k8]) = pb1;
      }
    } else {
      int k = t >> 3, n8 = (t & 7) << 3;
      #pragma unroll
      for (int i = 0; i < 8; ++i) {
        Bs[bufi][(n8 + i) * ASTR + k]      = pb0[i];
        Bs[bufi][(n8 + 64 + i) * ASTR + k] = pb1[i];
      }
    }
  };

  // ---- A fragment: row m; elems = K 16q..16q+15.
  //      B frag: col n; elems = K 16q..16q+15.
  auto ldAfrag = [&](const bf16* rowp) {
    v8bf lo = *(const v8bf*)(rowp);
    v8bf hi = *(const v8bf*)(rowp + 8);
    v16bf r;
    #pragma unroll
    for (int i = 0; i < 8; ++i) { r[i] = lo[i]; r[8 + i] = hi[i]; }
    return r;
  };

  // ---- prologue: stage K-tile 0 into buffer 0
  issue_async(0, 0);
  fetch(0);
  stage(0);
  int buf = 0;
  for (int kt = 0; kt < K; kt += 32) {
    const bool more = (kt + 32 < K);
    if (more) fetch(kt + 32);            // reg loads overlap with WMMAs below
    wait_async();                        // async writes into `buf` complete
    __syncthreads();                     // buf published; buf^1 reads retired
    if (more) issue_async(kt + 32, buf ^ 1);
    if (kt + 64 < K) prefetch(kt + 64);

    v16bf a0 = ldAfrag(&As[buf][(wm + fr) * ASTR + hq * 16]);
    v16bf a1 = ldAfrag(&As[buf][(wm + 16 + fr) * ASTR + hq * 16]);
    v16bf b0 = *(const v16bf*)(&Bs[buf][(wn + fr) * ASTR + hq * 16]);
    v16bf b1 = *(const v16bf*)(&Bs[buf][(wn + 16 + fr) * ASTR + hq * 16]);

    acc00 = __builtin_amdgcn_wmma_f32_16x16x32_bf16(false, a0, false, b0,
                                                    (short)0, acc00, false, false);
    acc01 = __builtin_amdgcn_wmma_f32_16x16x32_bf16(false, a0, false, b1,
                                                    (short)0, acc01, false, false);
    acc10 = __builtin_amdgcn_wmma_f32_16x16x32_bf16(false, a1, false, b0,
                                                    (short)0, acc10, false, false);
    acc11 = __builtin_amdgcn_wmma_f32_16x16x32_bf16(false, a1, false, b1,
                                                    (short)0, acc11, false, false);

    if (more) stage(buf ^ 1);            // reg-path ds stores into other buffer
    buf ^= 1;
  }

  // ---- epilogue: D VGPR v -> row m = v + 8*hq, col = fr (ISA C/D layout)
  const int rb = m0 + wm + hq * 8;
  const int c0 = n0 + wn + fr;
  const int c1 = c0 + 16;
  float bi0 = 0.f, bi1 = 0.f;
  if (bias) { bi0 = bias[c0]; bi1 = bias[c1]; }
  #pragma unroll
  for (int mi = 0; mi < 2; ++mi) {
    const v8f& d0 = mi ? acc10 : acc00;
    const v8f& d1 = mi ? acc11 : acc01;
    #pragma unroll
    for (int v = 0; v < 8; ++v) {
      int r = rb + mi * 16 + v;
      float x0 = d0[v] + bi0;
      float x1 = d1[v] + bi1;
      if (RELU) { x0 = fmaxf(x0, 0.f); x1 = fmaxf(x1, 0.f); }
      size_t i0 = (size_t)r * N + c0, i1 = (size_t)r * N + c1;
      if (OUTM == 0)      { outF[i0] = x0;          outF[i1] = x1; }
      else if (OUTM == 1) { outH[i0] = to_bf16(x0); outH[i1] = to_bf16(x1); }
      else                { outF[i0] += alpha * x0; outF[i1] += alpha * x1; }
    }
  }
}

// ---------------------------------------------------------------------------

extern "C" void kernel_launch(void* const* d_in, const int* in_sizes, int n_in,
                              void* d_out, int out_size, void* d_ws, size_t ws_size,
                              hipStream_t stream) {
  const float* x    = (const float*)d_in[0];  // (1024,1024)
  const float* W    = (const float*)d_in[1];  // (4096,4096)
  const float* adj  = (const float*)d_in[2];  // (4096,4096)
  const float* in_w = (const float*)d_in[3];  // (4096,1024)
  const float* in_b = (const float*)d_in[4];  // (4096,)
  const float* ow   = (const float*)d_in[5];  // (256,4096)
  const float* ob   = (const float*)d_in[6];  // (256,)

  const int Bt = 1024, DIN = 1024, Nn = 4096, DOUT = 256;

  float* outF   = (float*)d_out;                  // 1024*256
  float* credit = outF + (size_t)Bt * DOUT;       // 4096*4096

  char* ws = (char*)d_ws;
  bf16* Aw    = (bf16*)(ws);                      // 4096*4096  (32 MB)
  bf16* inwb  = (bf16*)(ws + 33554432);           // 4096*1024  ( 8 MB)
  bf16* outwb = (bf16*)(ws + 41943040);           // 256*4096   ( 2 MB)
  bf16* xb    = (bf16*)(ws + 44040192);           // 1024*1024  ( 2 MB)
  bf16* hA    = (bf16*)(ws + 46137344);           // 1024*4096  ( 8 MB)
  bf16* hB    = (bf16*)(ws + 54525952);           // 1024*4096  ( 8 MB)

  mask_mul_bf16<<<(Nn * Nn) / 256, 256, 0, stream>>>(W, adj, Aw, Nn * Nn);
  cvt_f32_bf16 <<<(Nn * DIN) / 256, 256, 0, stream>>>(in_w, inwb, Nn * DIN);
  cvt_f32_bf16 <<<(DOUT * Nn) / 256, 256, 0, stream>>>(ow, outwb, DOUT * Nn);
  cvt_f32_bf16 <<<(Bt * DIN) / 256, 256, 0, stream>>>(x, xb, Bt * DIN);
  zero_f32     <<<(Nn * Nn) / 256, 256, 0, stream>>>(credit, Nn * Nn);

  // h0 = relu(x @ in_w^T + in_b)          1024x1024 @ (1024x4096 via B^T)
  gemm_bf16_wmma<0, 1, 1, 1><<<dim3(Nn / 128, Bt / 64), 256, 0, stream>>>(
      xb, DIN, inwb, DIN, in_b, nullptr, hA, 0.f, Nn, DIN);

  bf16* hp = hA; bf16* hc = hB;
  for (int it = 0; it < 5; ++it) {
    // h_cur = relu(h_prev @ Aw)           1024x4096 @ 4096x4096
    gemm_bf16_wmma<0, 0, 1, 1><<<dim3(Nn / 128, Bt / 64), 256, 0, stream>>>(
        hp, Nn, Aw, Nn, nullptr, nullptr, hc, 0.f, Nn, Nn);
    // credit += h_prev^T @ h_cur / 1024   4096x1024 @ 1024x4096
    gemm_bf16_wmma<1, 0, 0, 2><<<dim3(Nn / 128, Nn / 64), 256, 0, stream>>>(
        hp, Nn, hc, Nn, nullptr, credit, nullptr, 1.0f / 1024.0f, Nn, Bt);
    bf16* tmp = hp; hp = hc; hc = tmp;
  }

  // out = h @ out_w^T + out_b             1024x4096 @ 4096x256
  gemm_bf16_wmma<0, 1, 0, 0><<<dim3(DOUT / 128, Bt / 64), 256, 0, stream>>>(
      hp, Nn, outwb, Nn, ob, outF, nullptr, 0.f, DOUT, Nn);
}
